// EMAU_12601434046672
// MI455X (gfx1250) — matrix operand order repaired and checked by reading, not compiled, not run on hardware
//
#include <hip/hip_runtime.h>
#include <hip/hip_bf16.h>

typedef __attribute__((ext_vector_type(16))) __bf16 v16bf;
typedef __attribute__((ext_vector_type(8)))  __bf16 v8bf;
typedef __attribute__((ext_vector_type(4)))  __bf16 v4bf;
typedef __attribute__((ext_vector_type(8)))  float  v8f;
typedef __attribute__((ext_vector_type(4)))  float  v4f;

#define BM 128
#define BN 64
#define BK 32
#define TILE_A (BM * BK)
#define TILE_B (BN * BK)

static __device__ __forceinline__ v4bf cvt4(v4f v) {
    v4bf r;
    r[0] = (__bf16)v[0]; r[1] = (__bf16)v[1];
    r[2] = (__bf16)v[2]; r[3] = (__bf16)v[3];
    return r;
}

// CDNA5 async copy: 16B global -> LDS per lane, tracked by ASYNCcnt.
static __device__ __forceinline__ void async_copy_b128(const void* g, unsigned lds_byte) {
    asm volatile("global_load_async_to_lds_b128 %0, %1, off"
                 :: "v"(lds_byte), "v"(g) : "memory");
}
static __device__ __forceinline__ void wait_async0() {
#if __has_builtin(__builtin_amdgcn_s_wait_asynccnt)
    __builtin_amdgcn_s_wait_asynccnt(0);
#else
    asm volatile("s_wait_asynccnt 0x0" ::: "memory");
#endif
}

// ---------------------------------------------------------------------------
// bf16 WMMA GEMM, f32 accumulate, double-buffered LDS.
//   - bf16 operands are staged with global_load_async_to_lds_b128 (raw copy;
//     layouts arranged so global-contig == LDS-contig).
//   - f32 operands are register-staged (v4f b128 loads) + converted to bf16.
// A is always non-transposed: element (m,k) = A[m*lda + k].
// B: transB ? (k,n)=B[n*ldb+k] : (k,n)=B[k*ldb+n].  bf16 B must be transB.
// ep_mode:
//   1: +bias ev[row]; bf16 store C0[row][col] + packed bf16 store C1[col][row]
//   2: relu; packed bf16 store C1[col][row]
//   3: *= 1/(1e-6+ev[col]); f32 C0[row][col] (atomicAdd when ksplit>1)
//   4: row-softmax over N==64; bf16 C0 + packed bf16 C1; colsum atomics -> eo
//   5: packed f32 store C1[col][row]; per-col sum/sumsq atomics -> eo (BN)
// strideC0/strideC1 are in BYTES; ep_stride, lda/ldb/ldc in elements.
// ---------------------------------------------------------------------------
template <typename TA, typename TB>
__global__ __launch_bounds__(256) void gemm_wmma(
    const TA* __restrict__ A, long lda, long strideA,
    const TB* __restrict__ Bg, long ldb, long strideB, int transB,
    void* __restrict__ C0v, long ldc0, long strideC0,
    void* __restrict__ C1v, long ldc1, long strideC1,
    int M, int N, int Kd, int n_tiles, int k_chunk,
    int ep_mode, const float* __restrict__ ep_vec, float* __restrict__ ep_out,
    long ep_stride)
{
    constexpr bool A_BF = (sizeof(TA) == 2);
    constexpr bool B_BF = (sizeof(TB) == 2);

    __shared__ __attribute__((aligned(32))) __bf16 sA[2 * TILE_A]; // [m][k]
    __shared__ __attribute__((aligned(32))) __bf16 sB[2 * TILE_B]; // [n][k]
    __shared__ float cs[64], cs2[64];

    const int b = blockIdx.z;
    A  += (long)b * strideA;
    Bg += (long)b * strideB;
    char* C0p = (char*)C0v + (long)b * strideC0;
    char* C1p = (char*)C1v + (long)b * strideC1;
    const float* ev = ep_vec ? (ep_vec + (long)b * ep_stride) : nullptr;
    float*       eo = ep_out ? (ep_out + (long)b * ep_stride) : nullptr;

    const int nb = (blockIdx.x % n_tiles) * BN;
    const int ks = blockIdx.x / n_tiles;
    const int ksplit = gridDim.x / n_tiles;
    const int k_begin = ks * k_chunk;
    const int k_end = k_begin + k_chunk;
    const int mb = blockIdx.y * BM;
    const int t = threadIdx.x;
    const int lane = t & 31;
    const int wave = t >> 5;
    const int lan16 = lane & 15;
    const int halfsel = lane >> 4;

    if (t < 64) { cs[t] = 0.0f; cs2[t] = 0.0f; }

    const unsigned sA_bytes = (unsigned)(unsigned long long)(void*)&sA[0];
    const unsigned sB_bytes = (unsigned)(unsigned long long)(void*)&sB[0];

    // ---------------- A staging setup (non-transposed, k-contiguous) --------
    const TA* gA = nullptr; long gA_inner = 0; int aoff = 0;     // async path
    const float* pa = nullptr; long innerA = 0; int sa_base = 0; // f32 path
    v4f ra[4]; v4f rb[2];
    if constexpr (A_BF) {
        gA = A + (long)(mb + (t >> 2)) * lda + k_begin + ((t & 3) << 3);
        gA_inner = 64 * lda;                          // +i*64 rows, i in {0,1}
        aoff = (t >> 2) * BK + ((t & 3) << 3);        // elems; +i*64*BK
    } else {
        pa = (const float*)A + (long)(mb + (t >> 3)) * lda + k_begin + ((t & 7) << 2);
        innerA = 32 * lda;
        sa_base = (t >> 3) * BK + ((t & 7) << 2);     // +i*32*BK, contig v4bf
    }
    // ---------------- B staging setup ---------------------------------------
    const TB* gB = nullptr; int boff = 0;                         // async path
    const float* pb = nullptr; long innerB = 0, stepB = 0; int sb_base = 0;
    if constexpr (B_BF) { // always transB-style: (k,n) = B[n*ldb + k]
        gB = Bg + (long)(nb + (t >> 2)) * ldb + k_begin + ((t & 3) << 3);
        boff = (t >> 2) * BK + ((t & 3) << 3);
    } else if (!transB) { // (k,n) = B[k*ldb + n]; 4 contiguous n per thread
        pb = (const float*)Bg + (long)(k_begin + (t >> 4)) * ldb + nb + ((t & 15) << 2);
        innerB = 16 * ldb; stepB = (long)BK * ldb;
        sb_base = ((t & 15) << 2) * BK + (t >> 4);    // scatter +j*BK +i*16
    } else {              // (k,n) = B[n*ldb + k]; 4 contiguous k per thread
        pb = (const float*)Bg + (long)(nb + (t >> 3)) * ldb + k_begin + ((t & 7) << 2);
        innerB = 32 * ldb; stepB = BK;
        sb_base = (t >> 3) * BK + ((t & 7) << 2);     // contig v4bf +i*32*BK
    }

    v8f acc[4] = {};

    // ---------------- prologue: first tile into buffer 0 --------------------
    if constexpr (A_BF) {
        async_copy_b128(gA,            sA_bytes + (unsigned)(aoff) * 2);
        async_copy_b128(gA + gA_inner, sA_bytes + (unsigned)(aoff + 64 * BK) * 2);
    } else {
        #pragma unroll
        for (int i = 0; i < 4; ++i) ra[i] = *(const v4f*)(pa + i * innerA);
    }
    if constexpr (B_BF) {
        async_copy_b128(gB, sB_bytes + (unsigned)(boff) * 2);
    } else {
        #pragma unroll
        for (int i = 0; i < 2; ++i) rb[i] = *(const v4f*)(pb + i * innerB);
    }

    int buf = 0;
    for (int k0 = k_begin; k0 < k_end; k0 += BK) {
        // register-path: convert + store current tile into sX[buf]
        if constexpr (!A_BF) {
            #pragma unroll
            for (int i = 0; i < 4; ++i)
                *(v4bf*)&sA[buf * TILE_A + sa_base + i * 32 * BK] = cvt4(ra[i]);
        }
        if constexpr (!B_BF) {
            if (!transB) {
                #pragma unroll
                for (int i = 0; i < 2; ++i)
                    #pragma unroll
                    for (int j = 0; j < 4; ++j)
                        sB[buf * TILE_B + sb_base + j * BK + i * 16] = (__bf16)rb[i][j];
            } else {
                #pragma unroll
                for (int i = 0; i < 2; ++i)
                    *(v4bf*)&sB[buf * TILE_B + sb_base + i * 32 * BK] = cvt4(rb[i]);
            }
        }
        if constexpr (A_BF || B_BF) wait_async0(); // own async writes to buf done
        __syncthreads();                           // everyone's tile visible

        // issue next tile into buf^1 (async engine / global loads run under WMMA)
        const bool next = (k0 + BK) < k_end;
        if (next) {
            if constexpr (A_BF) {
                gA += BK;
                async_copy_b128(gA,
                    sA_bytes + (unsigned)((buf ^ 1) * TILE_A + aoff) * 2);
                async_copy_b128(gA + gA_inner,
                    sA_bytes + (unsigned)((buf ^ 1) * TILE_A + aoff + 64 * BK) * 2);
            } else {
                pa += BK;
                #pragma unroll
                for (int i = 0; i < 4; ++i) ra[i] = *(const v4f*)(pa + i * innerA);
            }
            if constexpr (B_BF) {
                gB += BK;
                async_copy_b128(gB,
                    sB_bytes + (unsigned)((buf ^ 1) * TILE_B + boff) * 2);
            } else {
                pb += stepB;
                #pragma unroll
                for (int i = 0; i < 2; ++i) rb[i] = *(const v4f*)(pb + i * innerB);
            }
        }

        // ---- fragments (contiguous k = 16*half + s in both A and B) + 4 WMMAs ----
        union { v16bf v; v8bf h[2]; } fa;
        const __bf16* ap = &sA[buf * TILE_A + (wave * 16 + lan16) * BK + halfsel * 16];
        fa.h[0] = *(const v8bf*)(ap);
        fa.h[1] = *(const v8bf*)(ap + 8);
        v16bf fb[4];
        #pragma unroll
        for (int j = 0; j < 4; ++j)
            fb[j] = *(const v16bf*)&sB[buf * TILE_B + (j * 16 + lan16) * BK + halfsel * 16];
        #pragma unroll
        for (int j = 0; j < 4; ++j)
            acc[j] = __builtin_amdgcn_wmma_f32_16x16x32_bf16(
                false, fa.v, false, fb[j], (short)0, acc[j], false, false);
        buf ^= 1;
    }

    // ---------------- epilogues --------------------------------------------
    const int row_base = mb + wave * 16 + halfsel * 8;

    if (ep_mode == 4) {
        // softmax over the full 64-column row (this block owns all of N)
        #pragma unroll
        for (int r = 0; r < 8; ++r) {
            float mx = fmaxf(fmaxf(acc[0][r], acc[1][r]), fmaxf(acc[2][r], acc[3][r]));
            #pragma unroll
            for (int off = 8; off > 0; off >>= 1) mx = fmaxf(mx, __shfl_xor(mx, off));
            float s = 0.0f;
            #pragma unroll
            for (int j = 0; j < 4; ++j) { float e = __expf(acc[j][r] - mx); acc[j][r] = e; s += e; }
            #pragma unroll
            for (int off = 8; off > 0; off >>= 1) s += __shfl_xor(s, off);
            const float inv = 1.0f / s;
            #pragma unroll
            for (int j = 0; j < 4; ++j) acc[j][r] *= inv;
        }
        __bf16* O0 = (__bf16*)C0p; __bf16* O1 = (__bf16*)C1p;
        #pragma unroll
        for (int j = 0; j < 4; ++j) {
            const int col = nb + j * 16 + lan16;
            v8bf pk; float cp = 0.0f;
            #pragma unroll
            for (int r = 0; r < 8; ++r) {
                const float v = acc[j][r];
                O0[(long)(row_base + r) * ldc0 + col] = (__bf16)v;
                pk[r] = (__bf16)v; cp += v;
            }
            *(v8bf*)&O1[(long)col * ldc1 + row_base] = pk;  // z^T, packed
            cp += __shfl_xor(cp, 16);
            if (halfsel == 0) atomicAdd(&cs[j * 16 + lan16], cp);
        }
        __syncthreads();
        if (t < 64) atomicAdd(&eo[t], cs[t]);
        return;
    }

    if (ep_mode == 5) {
        // transposed packed f32 store + per-column BN sum/sumsq
        float* O1 = (float*)C1p;
        #pragma unroll
        for (int j = 0; j < 4; ++j) {
            const int col = nb + j * 16 + lan16;
            v4f lo, hi; float cp = 0.0f, cq = 0.0f;
            #pragma unroll
            for (int r = 0; r < 4; ++r) { lo[r] = acc[j][r]; hi[r] = acc[j][r + 4]; }
            #pragma unroll
            for (int r = 0; r < 8; ++r) { const float v = acc[j][r]; cp += v; cq += v * v; }
            float* dst = &O1[(long)col * ldc1 + row_base];
            *(v4f*)dst = lo; *(v4f*)(dst + 4) = hi;
            cp += __shfl_xor(cp, 16); cq += __shfl_xor(cq, 16);
            if (halfsel == 0) {
                atomicAdd(&cs[j * 16 + lan16], cp);
                atomicAdd(&cs2[j * 16 + lan16], cq);
            }
        }
        __syncthreads();
        if (t < 64) {
            atomicAdd(&eo[nb + t], cs[t]);
            atomicAdd(&eo[N + nb + t], cs2[t]);
        }
        return;
    }

    if (ep_mode == 1) { // bias + dual bf16 store (normal + transposed packed)
        __bf16* O0 = (__bf16*)C0p; __bf16* O1 = (__bf16*)C1p;
        #pragma unroll
        for (int j = 0; j < 4; ++j) {
            const int col = nb + j * 16 + lan16;
            v8bf pk;
            #pragma unroll
            for (int r = 0; r < 8; ++r) {
                const float v = acc[j][r] + ev[row_base + r];
                O0[(long)(row_base + r) * ldc0 + col] = (__bf16)v;
                pk[r] = (__bf16)v;
            }
            *(v8bf*)&O1[(long)col * ldc1 + row_base] = pk;
        }
        return;
    }

    if (ep_mode == 2) { // relu + transposed packed bf16 store
        __bf16* O1 = (__bf16*)C1p;
        #pragma unroll
        for (int j = 0; j < 4; ++j) {
            const int col = nb + j * 16 + lan16;
            v8bf pk;
            #pragma unroll
            for (int r = 0; r < 8; ++r) {
                float v = acc[j][r];
                pk[r] = (__bf16)(v > 0.0f ? v : 0.0f);
            }
            *(v8bf*)&O1[(long)col * ldc1 + row_base] = pk;
        }
        return;
    }

    // ep_mode 3 (and 0): f32 store / split-K atomic accumulate with col scale
    float* O0 = (float*)C0p;
    #pragma unroll
    for (int j = 0; j < 4; ++j) {
        const int col = nb + j * 16 + lan16;
        float scale = 1.0f;
        if (ep_mode == 3) scale = 1.0f / (1e-6f + ev[col]);
        #pragma unroll
        for (int r = 0; r < 8; ++r) {
            const float v = acc[j][r] * scale;
            if (ep_mode == 3 && ksplit > 1)
                atomicAdd(&O0[(long)(row_base + r) * ldc0 + col], v);
            else
                O0[(long)(row_base + r) * ldc0 + col] = v;
        }
    }
}

// ---------------------------------------------------------------------------
__global__ void bn_finalize(const float* __restrict__ accv,
                            float* __restrict__ meanv, float* __restrict__ rstdv,
                            int Cc, float cnt)
{
    const int c = blockIdx.x * blockDim.x + threadIdx.x;
    if (c < Cc) {
        const float mean = accv[c] / cnt;
        const float var = accv[Cc + c] / cnt - mean * mean;
        meanv[c] = mean;
        rstdv[c] = rsqrtf(var + 1e-5f);
    }
}

__global__ void bn_apply_residual(
    float* __restrict__ out, const float* __restrict__ x,
    const float* __restrict__ gamma, const float* __restrict__ beta,
    const float* __restrict__ meanv, const float* __restrict__ rstdv,
    long total, int Cc, int N)
{
    long i = (long)blockIdx.x * blockDim.x + threadIdx.x;
    const long stride = (long)gridDim.x * blockDim.x;
    for (; i < total; i += stride) {
        const int c = (int)((i / N) % Cc);
        const float v = out[i];
        out[i] = gamma[c] * (v - meanv[c]) * rstdv[c] + beta[c] + x[i];
    }
}

__global__ void init_mu(float* __restrict__ mu, const float* __restrict__ mu0,
                        long per_batch, long total)
{
    long i = (long)blockIdx.x * blockDim.x + threadIdx.x;
    if (i < total) mu[i] = mu0[i % per_batch];
}

__global__ void copy_mu(float* __restrict__ dst, const float* __restrict__ src, long n)
{
    long i = (long)blockIdx.x * blockDim.x + threadIdx.x;
    if (i < n) dst[i] = src[i];
}

// ---------------------------------------------------------------------------
extern "C" void kernel_launch(void* const* d_in, const int* in_sizes, int n_in,
                              void* d_out, int out_size, void* d_ws, size_t ws_size,
                              hipStream_t stream)
{
    (void)in_sizes; (void)n_in; (void)out_size; (void)ws_size;
    const float* x     = (const float*)d_in[0]; // (B,C,N)
    const float* w1    = (const float*)d_in[1]; // (C,C)
    const float* b1    = (const float*)d_in[2]; // (C)
    const float* mu0   = (const float*)d_in[3]; // (1,C,K)
    const float* w2    = (const float*)d_in[4]; // (C,C)
    const float* gamma = (const float*)d_in[5]; // (C)
    const float* beta  = (const float*)d_in[6]; // (C)
    float* out = (float*)d_out;                 // (B,C,N) ++ (B,C,K)

    const int B = 16, C = 512, K = 64, N = 8192;
    const long CN = (long)C * N, NK = (long)N * K, CK = (long)C * K;

    char* w = (char*)d_ws;
    __bf16* xs_bf  = (__bf16*)w; w += (size_t)B * CN * 2;  // (B,C,N)
    __bf16* xsT_bf = (__bf16*)w; w += (size_t)B * CN * 2;  // (B,N,C)
    __bf16* z_bf   = (__bf16*)w; w += (size_t)B * NK * 2;  // (B,N,K)
    __bf16* zT_bf  = (__bf16*)w; w += (size_t)B * NK * 2;  // (B,K,N)
    __bf16* x2T_bf = (__bf16*)w; w += (size_t)B * CN * 2;  // (B,N,C)
    float* mu      = (float*)w;  w += (size_t)B * CK * 4;  // (B,C,K)
    float* colsum  = (float*)w;  w += (size_t)B * K * 4;
    float* bnacc   = (float*)w;  w += (size_t)2 * C * 4;
    float* meanv   = (float*)w;  w += (size_t)C * 4;
    float* rstdv   = (float*)w;  w += (size_t)C * 4;

    const dim3 blk(256);

    { // mu <- broadcast(mu0)
        const long tot = (long)B * CK;
        init_mu<<<(unsigned)((tot + 255) / 256), blk, 0, stream>>>(mu, mu0, CK, tot);
    }

    // conv1: xs = w1 @ x + b1 ; writes xs (C,N) and xs^T (N,C) in bf16
    gemm_wmma<float, float><<<dim3(N / BN, C / BM, B), blk, 0, stream>>>(
        w1, C, 0, x, N, CN, /*transB*/0,
        xs_bf, N, CN * 2, xsT_bf, C, CN * 2,
        C, N, C, N / BN, C, /*ep*/1, b1, nullptr, 0);

    for (int stage = 0; stage < 3; ++stage) {
        hipMemsetAsync(colsum, 0, (size_t)B * K * sizeof(float), stream);
        // E-step: z = softmax_k(xs^T @ mu); writes z, z^T; colsum atomics
        gemm_wmma<__bf16, float><<<dim3(1, N / BM, B), blk, 0, stream>>>(
            xsT_bf, C, CN, mu, K, CK, /*transB*/0,
            z_bf, K, NK * 2, zT_bf, N, NK * 2,
            N, K, C, 1, C, /*ep*/4, nullptr, colsum, K);
        // M-step: mu = (xs @ z) * diag(1/(1e-6+colsum)), split-K x16 atomics
        hipMemsetAsync(mu, 0, (size_t)B * CK * sizeof(float), stream);
        gemm_wmma<__bf16, __bf16><<<dim3(16, C / BM, B), blk, 0, stream>>>(
            xs_bf, N, CN, zT_bf, N, NK, /*transB*/1,
            mu, K, CK * 4, nullptr, 0, 0,
            C, K, N, 1, N / 16, /*ep*/3, colsum, nullptr, K);
    }

    // recon: x2 = relu(mu @ z^T); stored transposed as x2^T (N,C) bf16
    gemm_wmma<float, __bf16><<<dim3(N / BN, C / BM, B), blk, 0, stream>>>(
        mu, K, CK, z_bf, K, NK, /*transB*/1,
        nullptr, 0, 0, x2T_bf, C, CN * 2,
        C, N, K, N / BN, K, /*ep*/2, nullptr, nullptr, 0);

    // conv2 (transposed form): x3^T = x2^T @ w2^T -> out[c*N+n] f32 + BN stats
    hipMemsetAsync(bnacc, 0, (size_t)2 * C * sizeof(float), stream);
    gemm_wmma<__bf16, float><<<dim3(C / BN, N / BM, B), blk, 0, stream>>>(
        x2T_bf, C, CN, w2, C, 0, /*transB*/1,
        nullptr, 0, 0, out, N, CN * 4,
        N, C, C, C / BN, C, /*ep*/5, nullptr, bnacc, 0);

    // BN finalize + apply + residual (in place on d_out)
    bn_finalize<<<(C + 255) / 256, blk, 0, stream>>>(bnacc, meanv, rstdv, C,
                                                     (float)B * (float)N);
    const long total = (long)B * CN;
    bn_apply_residual<<<4096, blk, 0, stream>>>(out, x, gamma, beta, meanv, rstdv,
                                                total, C, N);
    copy_mu<<<(unsigned)(((long)B * CK + 255) / 256), blk, 0, stream>>>(
        out + total, mu, (long)B * CK);
}
